// CausalSelfAttention_73598559584536
// MI455X (gfx1250) — matrix operand order repaired and checked by reading, not compiled, not run on hardware
//
#include <hip/hip_runtime.h>
#include <hip/hip_bf16.h>

// ---------------- problem constants ----------------
constexpr int BATCH = 4;
constexpr int SEQ   = 2048;
constexpr int DIM   = 1024;           // D_IN == D_OUT
constexpr int NROWS = BATCH * SEQ;    // 8192 flattened rows

typedef __attribute__((ext_vector_type(16))) __bf16 v16bf;
typedef __attribute__((ext_vector_type(8)))  __bf16 v8bf;
typedef __attribute__((ext_vector_type(8)))  float  v8f;

// ---------------- CDNA5 async global->LDS copy (b128 per lane) ----------------
#if __has_builtin(__builtin_amdgcn_global_load_async_to_lds_b128)
#define HAVE_ASYNC_LDS 1
#endif

// exact pointee type per hipcc diagnostic: vector_size(16) int
typedef __attribute__((__vector_size__(4 * sizeof(int)))) int vi4;

__device__ __forceinline__ void async_copy_b128(const void* g, void* l) {
#ifdef HAVE_ASYNC_LDS
    __builtin_amdgcn_global_load_async_to_lds_b128(
        (__attribute__((address_space(1))) vi4*)(uintptr_t)g,
        (__attribute__((address_space(3))) vi4*)(uintptr_t)l,
        0, 0);
#else
    *(v8bf*)l = *(const v8bf*)g;
#endif
}

__device__ __forceinline__ void async_join() {
#ifdef HAVE_ASYNC_LDS
#if __has_builtin(__builtin_amdgcn_s_wait_asynccnt)
    __builtin_amdgcn_s_wait_asynccnt(0);
#else
    asm volatile("s_wait_asynccnt 0" ::: "memory");
#endif
#endif
}

// ---------------- fragment loaders (CDNA5 WMMA layouts) ----------------
// A matrix 16x32 bf16: lanes 0-15 / 16-31 both hold rows M=0..15.
// lane half 0 -> K {0..7, 16..23}, half 1 -> K {8..15, 24..31}.
__device__ __forceinline__ v16bf load_a_frag(const __bf16* base, int ld) {
    int lane = threadIdx.x & 31;
    int row  = lane & 15;
    int half = lane >> 4;
    const __bf16* p = base + (size_t)row * ld + half * 8;
    v8bf lo = *(const v8bf*)(p);
    v8bf hi = *(const v8bf*)(p + 16);
    return __builtin_shufflevector(lo, hi, 0,1,2,3,4,5,6,7,8,9,10,11,12,13,14,15);
}

// B matrix 32x16 bf16: lane n = column, lane half 0 -> K {0..7, 16..23}, half 1 -> K {8..15, 24..31}.
// 'base' points at element [k=0][n=0]; row n is contiguous along K, stride ld.
__device__ __forceinline__ v16bf load_b_frag(const __bf16* base, int ld) {
    int lane = threadIdx.x & 31;
    int n  = lane & 15;
    int kh = lane >> 4;
    const __bf16* p = base + (size_t)n * ld + kh * 8;
    v8bf lo = *(const v8bf*)(p);
    v8bf hi = *(const v8bf*)(p + 16);
    return __builtin_shufflevector(lo, hi, 0,1,2,3,4,5,6,7,8,9,10,11,12,13,14,15);
}

__device__ __forceinline__ v8f wmma_bf16(v16bf a, v16bf b, v8f c) {
    return __builtin_amdgcn_wmma_f32_16x16x32_bf16(false, a, false, b, (short)0, c,
                                                   false, false);
}

// ---------------- kernel 1: f32 -> bf16 convert ----------------
__global__ void cvt_f32_bf16(const float* __restrict__ src,
                             __bf16* __restrict__ dst, int n) {
    int i = (blockIdx.x * blockDim.x + threadIdx.x) * 8;
    if (i >= n) return;
    float4 f0 = *(const float4*)(src + i);
    float4 f1 = *(const float4*)(src + i + 4);
    v8bf o;
    o[0] = (__bf16)f0.x; o[1] = (__bf16)f0.y; o[2] = (__bf16)f0.z; o[3] = (__bf16)f0.w;
    o[4] = (__bf16)f1.x; o[5] = (__bf16)f1.y; o[6] = (__bf16)f1.z; o[7] = (__bf16)f1.w;
    *(v8bf*)(dst + i) = o;
}

// ---------------- kernel 2: Y = X * W^T (bf16 in, bf16 out) ----------------
// One block = full 16-row x 1024-col output band. X band (32KB) staged to LDS
// once via async-to-LDS and shared by all 8 waves (8x less A traffic).
// Each wave owns a 128-wide N slice: 8 WMMA per 32-wide k-step, with the next
// step's 8 B fragments double-buffered in registers to hide load latency.
__global__ void __launch_bounds__(256)
gemm_bf16_wmma(const __bf16* __restrict__ X,
               const __bf16* __restrict__ W,
               __bf16* __restrict__ Y,
               float scale, int transposeOut) {
    __shared__ __align__(16) __bf16 sX[16 * DIM];   // 32 KB

    int row0 = blockIdx.x * 16;
    int t    = threadIdx.x;

    // ---- stage X band into LDS: 256 threads x 8 x 16B = 32KB ----
    {
        const __bf16* g = X + (size_t)row0 * DIM;
#pragma unroll
        for (int i = 0; i < 8; ++i) {
            int off = t * 8 + i * 2048;             // elements; 16B per chunk
            async_copy_b128(g + off, &sX[off]);
        }
        async_join();
        __syncthreads();
    }

    int w    = t >> 5;
    int lane = t & 31;
    int col0 = w * 128;

    v8f acc[8];
#pragma unroll
    for (int j = 0; j < 8; ++j) acc[j] = (v8f){0,0,0,0,0,0,0,0};

    // ---- software-pipelined main loop ----
    v16bf bb[2][8];
#pragma unroll
    for (int j = 0; j < 8; ++j)
        bb[0][j] = load_b_frag(W + (size_t)(col0 + j * 16) * DIM, DIM);

    int p = 0;
    for (int k0 = 0; k0 < DIM; k0 += 32, p ^= 1) {
        int kn = (k0 + 32 < DIM) ? (k0 + 32) : k0;  // clamped prefetch
#pragma unroll
        for (int j = 0; j < 8; ++j)
            bb[p ^ 1][j] = load_b_frag(W + (size_t)(col0 + j * 16) * DIM + kn, DIM);

        // A fragment from LDS
        int arow = lane & 15, ahalf = lane >> 4;
        v8bf alo = *(const v8bf*)&sX[arow * DIM + k0 + ahalf * 8];
        v8bf ahi = *(const v8bf*)&sX[arow * DIM + k0 + 16 + ahalf * 8];
        v16bf a  = __builtin_shufflevector(alo, ahi,
                       0,1,2,3,4,5,6,7,8,9,10,11,12,13,14,15);

#pragma unroll
        for (int j = 0; j < 8; ++j)
            acc[j] = wmma_bf16(a, bb[p][j], acc[j]);
    }

    // ---- store ----
    int nl = lane & 15;
    int mh = (lane >> 4) * 8;
#pragma unroll
    for (int j = 0; j < 8; ++j) {
#pragma unroll
        for (int r = 0; r < 8; ++r) {
            float v = acc[j][r] * scale;
            int row = row0 + mh + r;
            int col = col0 + j * 16 + nl;
            if (!transposeOut) {
                Y[(size_t)row * DIM + col] = (__bf16)v;
            } else {
                int b = row >> 11;                  // / SEQ
                int s = row & (SEQ - 1);
                Y[((size_t)(b * DIM + col) << 11) + s] = (__bf16)v;
            }
        }
    }
}

// ---------------- kernel 3: causal flash attention ----------------
// One 16-row query tile per block (8 waves); waves split d into 128-wide slices.
// Key blocks of 32. Partial QK^T reduced across waves via LDS; softmax with
// lane shuffles + v_exp; P staged through LDS (C-layout -> A-layout) for P*V.
// All 8 B fragments batched ahead of each WMMA chain; next key block prefetched.
__global__ void __launch_bounds__(256)
attn_flash(const __bf16* __restrict__ Qb,   // [NROWS][DIM], pre-scaled by 1/32
           const __bf16* __restrict__ Kb,   // [NROWS][DIM]
           const __bf16* __restrict__ Vt,   // [BATCH][DIM][SEQ]
           float* __restrict__ out) {       // [NROWS][DIM] f32
    __shared__ __align__(16) float  sS[8][16 * 32];
    __shared__ __align__(16) __bf16 sP[8][16 * 32];

    int w    = threadIdx.x >> 5;
    int lane = threadIdx.x & 31;
    int nl   = lane & 15;
    int mh   = (lane >> 4) * 8;

    int b  = blockIdx.x >> 7;          // / (SEQ/16)
    int qt = blockIdx.x & 127;
    int q0 = qt * 16;
    int d0 = w * 128;                  // this wave's d slice

    // preload Q fragments for our d slice (4 x 16x32)
    const __bf16* Qbase = Qb + ((size_t)(b * SEQ + q0)) * DIM + d0;
    v16bf qf[4];
#pragma unroll
    for (int c = 0; c < 4; ++c) qf[c] = load_a_frag(Qbase + c * 32, DIM);

    v8f accO[8];
#pragma unroll
    for (int c = 0; c < 8; ++c) accO[c] = (v8f){0,0,0,0,0,0,0,0};
    float mrow[8], lrow[8];
#pragma unroll
    for (int r = 0; r < 8; ++r) { mrow[r] = -1e30f; lrow[r] = 0.0f; }

    int kbmax = (q0 + 15) >> 5;
    for (int kb = 0; kb <= kbmax; ++kb) {
        int key0 = kb * 32;
        const __bf16* Kbase = Kb + ((size_t)(b * SEQ + key0)) * DIM + d0;
        const __bf16* Vbase = Vt + ((size_t)(b * DIM + d0)) * SEQ + key0;

        // prefetch next key block (lowers to global_prefetch_b8)
        if (kb < kbmax) {
            __builtin_prefetch(Kbase + (size_t)32 * DIM, 0, 1);
            __builtin_prefetch(Vbase + 32, 0, 1);
        }

        // --- batch-load all 8 K fragments, then 8 WMMAs (16x32 scores) ---
        v16bf kf[8];
#pragma unroll
        for (int c = 0; c < 4; ++c) {
            kf[c]     = load_b_frag(Kbase + c * 32, DIM);                    // keys 0..15
            kf[4 + c] = load_b_frag(Kbase + (size_t)16 * DIM + c * 32, DIM); // keys 16..31
        }
        v8f s0 = (v8f){0,0,0,0,0,0,0,0};
        v8f s1 = (v8f){0,0,0,0,0,0,0,0};
#pragma unroll
        for (int c = 0; c < 4; ++c) {
            s0 = wmma_bf16(qf[c], kf[c],     s0);
            s1 = wmma_bf16(qf[c], kf[4 + c], s1);
        }

        // --- cross-wave reduction of partial scores through LDS ---
#pragma unroll
        for (int r = 0; r < 8; ++r) {
            sS[w][(mh + r) * 32 + nl]      = s0[r];
            sS[w][(mh + r) * 32 + 16 + nl] = s1[r];
        }
        __syncthreads();
        float p0[8], p1[8];
#pragma unroll
        for (int r = 0; r < 8; ++r) {
            float v0 = 0.0f, v1 = 0.0f;
#pragma unroll
            for (int w2 = 0; w2 < 8; ++w2) {
                v0 += sS[w2][(mh + r) * 32 + nl];
                v1 += sS[w2][(mh + r) * 32 + 16 + nl];
            }
            p0[r] = v0; p1[r] = v1;
        }
        __syncthreads();   // before next iteration's writes

        // --- causal mask + online softmax ---
#pragma unroll
        for (int r = 0; r < 8; ++r) {
            int q = q0 + mh + r;
            p0[r] = (key0 + nl      > q) ? -1e30f : p0[r];
            p1[r] = (key0 + 16 + nl > q) ? -1e30f : p1[r];
            float mloc = fmaxf(p0[r], p1[r]);
#pragma unroll
            for (int msk = 1; msk <= 8; msk <<= 1)
                mloc = fmaxf(mloc, __shfl_xor(mloc, msk, 32));
            float mnew  = fmaxf(mrow[r], mloc);
            float alpha = __expf(mrow[r] - mnew);
            mrow[r] = mnew;
            p0[r] = __expf(p0[r] - mnew);
            p1[r] = __expf(p1[r] - mnew);
            float psum = p0[r] + p1[r];
#pragma unroll
            for (int msk = 1; msk <= 8; msk <<= 1)
                psum += __shfl_xor(psum, msk, 32);
            lrow[r] = lrow[r] * alpha + psum;
#pragma unroll
            for (int c = 0; c < 8; ++c) accO[c][r] *= alpha;
        }

        // --- stage P (bf16) through LDS to get the A-matrix layout ---
#pragma unroll
        for (int r = 0; r < 8; ++r) {
            sP[w][(mh + r) * 32 + nl]      = (__bf16)p0[r];
            sP[w][(mh + r) * 32 + 16 + nl] = (__bf16)p1[r];
        }
        // same-wave LDS RAW: compiler inserts s_wait_dscnt
        int arow = lane & 15, ahalf = lane >> 4;
        v8bf plo = *(const v8bf*)&sP[w][arow * 32 + ahalf * 8];
        v8bf phi = *(const v8bf*)&sP[w][arow * 32 + 16 + ahalf * 8];
        v16bf pa = __builtin_shufflevector(plo, phi,
                     0,1,2,3,4,5,6,7,8,9,10,11,12,13,14,15);

        // --- batch-load all 8 V fragments, then 8 WMMAs (accO += P*V) ---
        v16bf vf[8];
#pragma unroll
        for (int c = 0; c < 8; ++c)
            vf[c] = load_b_frag(Vbase + (size_t)(c * 16) * SEQ, SEQ);
#pragma unroll
        for (int c = 0; c < 8; ++c)
            accO[c] = wmma_bf16(pa, vf[c], accO[c]);
    }

    // --- epilogue: normalize and store f32 ---
#pragma unroll
    for (int r = 0; r < 8; ++r) {
        float invl = 1.0f / lrow[r];
        size_t rowbase = ((size_t)(b * SEQ + q0 + mh + r)) * DIM + d0;
#pragma unroll
        for (int c = 0; c < 8; ++c)
            out[rowbase + c * 16 + nl] = accO[c][r] * invl;
    }
}

// ---------------- host launcher ----------------
extern "C" void kernel_launch(void* const* d_in, const int* in_sizes, int n_in,
                              void* d_out, int out_size, void* d_ws, size_t ws_size,
                              hipStream_t stream) {
    const float* x  = (const float*)d_in[0];
    const float* Wq = (const float*)d_in[1];
    const float* Wk = (const float*)d_in[2];
    const float* Wv = (const float*)d_in[3];
    float* out = (float*)d_out;

    const size_t NX = (size_t)NROWS * DIM;   // 8,388,608
    const size_t NW = (size_t)DIM * DIM;     // 1,048,576

    char* ws = (char*)d_ws;
    __bf16* xb  = (__bf16*)(ws);
    __bf16* wqb = (__bf16*)(ws + NX * 2);
    __bf16* wkb = (__bf16*)(ws + NX * 2 + NW * 2);
    __bf16* wvb = (__bf16*)(ws + NX * 2 + NW * 4);
    __bf16* qb  = (__bf16*)(ws + NX * 2 + NW * 6);
    __bf16* kbf = (__bf16*)(ws + NX * 4 + NW * 6);
    __bf16* vtb = (__bf16*)(ws + NX * 6 + NW * 6);

    // 1) f32 -> bf16
    cvt_f32_bf16<<<(int)(NX / (256 * 8)), 256, 0, stream>>>(x,  xb,  (int)NX);
    cvt_f32_bf16<<<(int)(NW / (256 * 8)), 256, 0, stream>>>(Wq, wqb, (int)NW);
    cvt_f32_bf16<<<(int)(NW / (256 * 8)), 256, 0, stream>>>(Wk, wkb, (int)NW);
    cvt_f32_bf16<<<(int)(NW / (256 * 8)), 256, 0, stream>>>(Wv, wvb, (int)NW);

    // 2) QKV projections (Q pre-scaled by 1/sqrt(D) = 1/32; V stored transposed)
    gemm_bf16_wmma<<<NROWS / 16, 256, 0, stream>>>(xb, wqb, qb,  1.0f / 32.0f, 0);
    gemm_bf16_wmma<<<NROWS / 16, 256, 0, stream>>>(xb, wkb, kbf, 1.0f,         0);
    gemm_bf16_wmma<<<NROWS / 16, 256, 0, stream>>>(xb, wvb, vtb, 1.0f,         1);

    // 3) causal flash attention: one 16-row query tile per block
    attn_flash<<<BATCH * (SEQ / 16), 256, 0, stream>>>(qb, kbf, vtb, out);
}
